// StabilityPredictorLA_43009802502317
// MI455X (gfx1250) — matrix operand order repaired and checked by reading, not compiled, not run on hardware
//
#include <hip/hip_runtime.h>
#include <hip/hip_bf16.h>
#include <cstddef>

// ---------------- problem constants ----------------
constexpr int Bc = 8, Lc = 512, Kc = 48, Hc = 128;
constexpr int NNc = Bc * Lc;        // 4096 nodes
constexpr int NEc = NNc * Kc;       // 196608 edges
constexpr int EDGE_IN = 416;

// ---------------- workspace layout (float units) ----------------
constexpr size_t o_atoms = 0;                                  // 5*NN*3
constexpr size_t o_matt  = o_atoms + (size_t)5 * NNc * 3;      // NE
constexpr size_t o_Dnb   = o_matt + (size_t)NEc;               // NE
constexpr size_t o_Eidx  = o_Dnb + (size_t)NEc;                // NE (int)
constexpr size_t o_rank  = o_Eidx + (size_t)NEc;               // NN (int)
constexpr size_t o_hV    = o_rank + (size_t)NNc;               // NN*H
constexpr size_t o_hVenc = o_hV + (size_t)NNc * Hc;            // NN*H
constexpr size_t o_hS    = o_hVenc + (size_t)NNc * Hc;         // NN*H
constexpr size_t o_nodeT = o_hS + (size_t)NNc * Hc;            // NN*H
constexpr size_t o_ffn   = o_nodeT + (size_t)NNc * Hc;         // NN*512
constexpr size_t o_o     = o_ffn + (size_t)NNc * 512;          // B*H*L
constexpr size_t o_lg    = o_o + (size_t)Bc * Hc * Lc;         // B*H*L
constexpr size_t o_gf    = o_lg + (size_t)Bc * Hc * Lc;        // B*H
constexpr size_t o_h1    = o_gf + (size_t)Bc * Hc;             // B*H
constexpr size_t o_h2    = o_h1 + (size_t)Bc * Hc;             // B*64
constexpr size_t o_hE    = o_h2 + (size_t)Bc * 64;             // NE*H
constexpr size_t o_m1    = o_hE + (size_t)NEc * Hc;            // NE*H
constexpr size_t o_m2    = o_m1 + (size_t)NEc * Hc;            // NE*H
constexpr size_t o_hES   = o_m2 + (size_t)NEc * Hc;            // NE*256
constexpr size_t o_hEXV  = o_hES + (size_t)NEc * 256;          // NE*384
constexpr size_t o_big   = o_hEXV + (size_t)NEc * 384;         // NE*512 (Efeat / h_EV)

// ---------------- WMMA types ----------------
typedef __attribute__((ext_vector_type(16))) _Float16 v16h;
typedef __attribute__((ext_vector_type(8)))  float    v8f;

__device__ __forceinline__ float gelu_exact(float x) {
    return 0.5f * x * (1.0f + erff(x * 0.70710678118654752f));
}

// A fragment: relative K offsets {0..15} contiguous from base pointer
__device__ __forceinline__ v16h load_frag_a(const float* p) {
    const float4* q = (const float4*)p;
    float4 x0 = q[0], x1 = q[1], x2 = q[2], x3 = q[3];
    v16h r;
    r[0]  = (_Float16)x0.x; r[1]  = (_Float16)x0.y;
    r[2]  = (_Float16)x0.z; r[3]  = (_Float16)x0.w;
    r[4]  = (_Float16)x1.x; r[5]  = (_Float16)x1.y;
    r[6]  = (_Float16)x1.z; r[7]  = (_Float16)x1.w;
    r[8]  = (_Float16)x2.x; r[9]  = (_Float16)x2.y;
    r[10] = (_Float16)x2.z; r[11] = (_Float16)x2.w;
    r[12] = (_Float16)x3.x; r[13] = (_Float16)x3.y;
    r[14] = (_Float16)x3.z; r[15] = (_Float16)x3.w;
    return r;
}
// B fragment: relative K offsets {0..15} contiguous
__device__ __forceinline__ v16h load_frag_b(const float* p) {
    const float4* q = (const float4*)p;
    float4 x0 = q[0], x1 = q[1], x2 = q[2], x3 = q[3];
    v16h r;
    r[0]  = (_Float16)x0.x; r[1]  = (_Float16)x0.y;
    r[2]  = (_Float16)x0.z; r[3]  = (_Float16)x0.w;
    r[4]  = (_Float16)x1.x; r[5]  = (_Float16)x1.y;
    r[6]  = (_Float16)x1.z; r[7]  = (_Float16)x1.w;
    r[8]  = (_Float16)x2.x; r[9]  = (_Float16)x2.y;
    r[10] = (_Float16)x2.z; r[11] = (_Float16)x2.w;
    r[12] = (_Float16)x3.x; r[13] = (_Float16)x3.y;
    r[14] = (_Float16)x3.z; r[15] = (_Float16)x3.w;
    return r;
}

// =====================================================================
// Generic GEMM: Y[M,N] = act(X[M,Kd] @ W[N,Kd]^T + bias[N])
// grid(ceil(M/64), N/32), block 128 (4 waves).
// Each wave: one 16-row tile x two 16-col tiles (shared A fragment, two
// independent WMMA accumulator chains for ILP). Out-of-range A rows are
// CLAMPED to M-1 (their D rows are never stored), keeping control flow
// uniform and EXEC all-ones as WMMA requires.
// =====================================================================
__global__ void gemm_wmma_kernel(const float* __restrict__ X,
                                 const float* __restrict__ W,
                                 const float* __restrict__ bias,
                                 float* __restrict__ Y,
                                 int M, int N, int Kd, int act)
{
    const int wave = threadIdx.x >> 5;
    const int lane = threadIdx.x & 31;
    const int half = lane >> 4;
    const int l16  = lane & 15;
    const int m0 = blockIdx.x * 64 + wave * 16;
    const int n0 = blockIdx.y * 32;

    int arow = m0 + l16;
    if (arow >= M) arow = M - 1;   // clamp: garbage rows are never stored

    // A: lanes 0-15 K {0..15}; lanes 16-31 K {16..31}
    const float* Xr = X + (size_t)arow * Kd + 16 * half;
    // B[K,N] = W[N,K]: lane -> N = n0+l16 (+16), K = kc + 16*half + 0..15
    const float* W0 = W + (size_t)(n0 + l16) * Kd + 16 * half;
    const float* W1 = W0 + (size_t)16 * Kd;

    v8f acc0 = {}, acc1 = {};
    for (int kc = 0; kc < Kd; kc += 32) {
        v16h a  = load_frag_a(Xr + kc);
        v16h b0 = load_frag_b(W0 + kc);
        v16h b1 = load_frag_b(W1 + kc);
        acc0 = __builtin_amdgcn_wmma_f32_16x16x32_f16(
            false, a, false, b0, (short)0, acc0, false, false);
        acc1 = __builtin_amdgcn_wmma_f32_16x16x32_f16(
            false, a, false, b1, (short)0, acc1, false, false);
    }

    const float bn0 = bias ? bias[n0 + l16]      : 0.f;
    const float bn1 = bias ? bias[n0 + 16 + l16] : 0.f;
#pragma unroll
    for (int r = 0; r < 8; ++r) {
        int row = m0 + 8 * half + r;   // f32 D: VGPR r -> M = 8*half + r, N = l16
        if (row < M) {
            float y0 = acc0[r] + bn0;
            float y1 = acc1[r] + bn1;
            if (act == 1) { y0 = gelu_exact(y0); y1 = gelu_exact(y1); }
            Y[(size_t)row * N + (n0 + l16)]      = y0;
            Y[(size_t)row * N + (n0 + 16 + l16)] = y1;
        }
    }
}

// =====================================================================
// geometry: atoms[5][B*L][3] = {Ca, Nb, Cc, Oo, Cb}
// =====================================================================
__global__ void geom_kernel(const float* __restrict__ X, float* __restrict__ atoms)
{
    int idx = blockIdx.x * blockDim.x + threadIdx.x;
    if (idx >= NNc) return;
    const float* x = X + (size_t)idx * 12;
    float Nb[3], Ca[3], Cc[3], Oo[3], bv[3], cv[3], av[3], Cb[3];
#pragma unroll
    for (int c = 0; c < 3; ++c) { Nb[c] = x[c]; Ca[c] = x[3 + c]; Cc[c] = x[6 + c]; Oo[c] = x[9 + c]; }
#pragma unroll
    for (int c = 0; c < 3; ++c) { bv[c] = Ca[c] - Nb[c]; cv[c] = Cc[c] - Ca[c]; }
    av[0] = bv[1] * cv[2] - bv[2] * cv[1];
    av[1] = bv[2] * cv[0] - bv[0] * cv[2];
    av[2] = bv[0] * cv[1] - bv[1] * cv[0];
#pragma unroll
    for (int c = 0; c < 3; ++c)
        Cb[c] = -0.58273431f * av[c] + 0.56802827f * bv[c] - 0.54067466f * cv[c] + Ca[c];
#pragma unroll
    for (int c = 0; c < 3; ++c) {
        atoms[((size_t)0 * NNc + idx) * 3 + c] = Ca[c];
        atoms[((size_t)1 * NNc + idx) * 3 + c] = Nb[c];
        atoms[((size_t)2 * NNc + idx) * 3 + c] = Cc[c];
        atoms[((size_t)3 * NNc + idx) * 3 + c] = Oo[c];
        atoms[((size_t)4 * NNc + idx) * 3 + c] = Cb[c];
    }
}

// =====================================================================
// top-K=48 nearest per row via iterated argmin in LDS (L = 512)
// =====================================================================
__global__ void topk_kernel(const float* __restrict__ atoms, const float* __restrict__ mask,
                            int* __restrict__ Eidx, float* __restrict__ Dnb)
{
    int row = blockIdx.x;              // b*L+i
    int b   = row / Lc;
    int t   = threadIdx.x;             // 256 threads
    __shared__ float ds[Lc];
    __shared__ float m2s[Lc];
    __shared__ float rv[256];
    __shared__ int   ri[256];

    const float* ca = atoms + (size_t)row * 3;
    float cx = ca[0], cy = ca[1], cz = ca[2];
    float mi = mask[row];
    for (int j = t; j < Lc; j += 256) {
        const float* cj = atoms + (size_t)(b * Lc + j) * 3;
        float m2 = mi * mask[b * Lc + j];
        float dx = cx - cj[0], dy = cy - cj[1], dz = cz - cj[2];
        ds[j]  = m2 * sqrtf(dx * dx + dy * dy + dz * dz + 1e-6f);
        m2s[j] = m2;
    }
    __syncthreads();
    // row max
    rv[t] = fmaxf(ds[t], ds[t + 256]);
    __syncthreads();
    for (int s = 128; s >= 1; s >>= 1) {
        if (t < s) rv[t] = fmaxf(rv[t], rv[t + s]);
        __syncthreads();
    }
    float dmax = rv[0];
    __syncthreads();
    for (int j = t; j < Lc; j += 256) ds[j] = ds[j] + (1.f - m2s[j]) * dmax;
    __syncthreads();

    for (int it = 0; it < Kc; ++it) {
        float v0 = ds[t], v1 = ds[t + 256];
        float bv; int bi;
        if (v1 < v0) { bv = v1; bi = t + 256; } else { bv = v0; bi = t; }
        rv[t] = bv; ri[t] = bi;
        __syncthreads();
        for (int s = 128; s >= 1; s >>= 1) {
            if (t < s) {
                float ov = rv[t + s]; int oi = ri[t + s];
                if (ov < rv[t] || (ov == rv[t] && oi < ri[t])) { rv[t] = ov; ri[t] = oi; }
            }
            __syncthreads();
        }
        if (t == 0) {
            Eidx[(size_t)row * Kc + it] = ri[0];
            Dnb [(size_t)row * Kc + it] = rv[0];
            ds[ri[0]] = 3.0e38f;
        }
        __syncthreads();
    }
}

// =====================================================================
// mask_attend[e] = mask[b,j] * mask[b,i]
// =====================================================================
__global__ void matt_kernel(const float* __restrict__ mask, const int* __restrict__ Eidx,
                            float* __restrict__ matt)
{
    int e = blockIdx.x * blockDim.x + threadIdx.x;
    if (e >= NEc) return;
    int node = e / Kc, b = node / Lc;
    matt[e] = mask[b * Lc + Eidx[e]] * mask[node];
}

// =====================================================================
// edge features: 16 positional + 25 pairs x 16 RBF = 416
// =====================================================================
__constant__ int PAIR_A[25] = {0,1,2,3,4, 0,0,0,0, 1,1,1, 4,4,3, 1,2,3,4, 2,3,4, 2,3,2};
__constant__ int PAIR_B[25] = {0,1,2,3,4, 1,2,3,4, 2,3,4, 2,3,2, 0,0,0,0, 1,1,1, 4,4,3};

__global__ void edgefeat_kernel(const float* __restrict__ atoms, const int* __restrict__ Eidx,
                                const float* __restrict__ Dnb, const int* __restrict__ residx,
                                const int* __restrict__ chain, const float* __restrict__ posw,
                                const float* __restrict__ posb, float* __restrict__ Ef)
{
    int e = blockIdx.x * blockDim.x + threadIdx.x;
    if (e >= NEc) return;
    int node = e / Kc, b = node / Lc;
    int jg = b * Lc + Eidx[e];
    float* out = Ef + (size_t)e * EDGE_IN;

    int off  = residx[node] - residx[jg];
    int same = (chain[node] == chain[jg]) ? 1 : 0;
    int dc   = same ? min(max(off + 32, 0), 64) : 65;
#pragma unroll
    for (int f = 0; f < 16; ++f) out[f] = posw[f * 66 + dc] + posb[f];

    for (int p = 0; p < 25; ++p) {
        float d;
        if (p == 0) d = Dnb[e];
        else {
            int ia = PAIR_A[p], ib = PAIR_B[p];
            const float* A  = atoms + ((size_t)ia * NNc + node) * 3;
            const float* Bp = atoms + ((size_t)ib * NNc + jg) * 3;
            float dx = A[0] - Bp[0], dy = A[1] - Bp[1], dz = A[2] - Bp[2];
            d = sqrtf(dx * dx + dy * dy + dz * dz + 1e-6f);
        }
#pragma unroll
        for (int m = 0; m < 16; ++m) {
            float mu = 2.f + (20.f / 15.f) * (float)m;
            float z  = (d - mu) * (1.f / 1.25f);
            out[16 + p * 16 + m] = __expf(-z * z);
        }
    }
}

// =====================================================================
// layernorm over H=128 (one wave per row), fused residual + row mask
// =====================================================================
__global__ void ln_kernel(const float* __restrict__ Xin, const float* __restrict__ resid,
                          const float* __restrict__ g, const float* __restrict__ be,
                          const float* __restrict__ rowmask, float* __restrict__ Y, int rows)
{
    int wave = threadIdx.x >> 5, lane = threadIdx.x & 31;
    int row = blockIdx.x * 8 + wave;
    if (row >= rows) return;
    const float* xp = Xin + (size_t)row * Hc;
    const float* rp = resid ? resid + (size_t)row * Hc : nullptr;
    float v[4]; float s = 0.f, sq = 0.f;
#pragma unroll
    for (int t = 0; t < 4; ++t) {
        float x = xp[lane * 4 + t] + (rp ? rp[lane * 4 + t] : 0.f);
        v[t] = x; s += x; sq += x * x;
    }
    for (int off = 16; off >= 1; off >>= 1) {
        s  += __shfl_xor(s, off, 32);
        sq += __shfl_xor(sq, off, 32);
    }
    float mean = s * (1.f / 128.f);
    float var  = sq * (1.f / 128.f) - mean * mean;
    float inv  = rsqrtf(var + 1e-5f);
    float mk   = rowmask ? rowmask[row] : 1.f;
    float* yp  = Y + (size_t)row * Hc;
#pragma unroll
    for (int t = 0; t < 4; ++t)
        yp[lane * 4 + t] = ((v[t] - mean) * inv * g[lane * 4 + t] + be[lane * 4 + t]) * mk;
}

// =====================================================================
// concat [h_V[i] | h_E | h_V[j]] -> 384 per edge
// =====================================================================
__global__ void concat3_kernel(const float* __restrict__ hV, const float* __restrict__ hE,
                               const int* __restrict__ Eidx, float* __restrict__ out)
{
    int e = blockIdx.x, h = threadIdx.x;         // 384 threads
    int node = e / Kc, b = node / Lc;
    int jg = b * Lc + Eidx[e];
    float v;
    if (h < 128)      v = hV[(size_t)node * Hc + h];
    else if (h < 256) v = hE[(size_t)e * Hc + (h - 128)];
    else              v = hV[(size_t)jg * Hc + (h - 256)];
    out[(size_t)e * 384 + h] = v;
}

// =====================================================================
// message reduce: out = hV + (sum_k m*att) / 30
// =====================================================================
__global__ void msgred_kernel(const float* __restrict__ m, const float* __restrict__ att,
                              const float* __restrict__ hV, float* __restrict__ out)
{
    int node = blockIdx.x, h = threadIdx.x;      // 128 threads
    const float* mp = m + (size_t)node * Kc * Hc + h;
    float acc = 0.f;
    for (int k = 0; k < Kc; ++k)
        acc += mp[(size_t)k * Hc] * (att ? att[node * Kc + k] : 1.f);
    out[(size_t)node * Hc + h] = hV[(size_t)node * Hc + h] + acc * (1.f / 30.f);
}

__global__ void zero_kernel(float* __restrict__ p, int n)
{
    int i = blockIdx.x * blockDim.x + threadIdx.x;
    if (i < n) p[i] = 0.f;
}

// ---------------- decoder setup ----------------
__global__ void hs_kernel(const int* __restrict__ aa, const float* __restrict__ Ws,
                          float* __restrict__ hS)
{
    int node = blockIdx.x, h = threadIdx.x;
    hS[(size_t)node * Hc + h] = Ws[(size_t)aa[node] * Hc + h];
}

__global__ void rank_kernel(const float* __restrict__ mask, const float* __restrict__ chainM,
                            const float* __restrict__ drand, int* __restrict__ rank)
{
    int row = blockIdx.x * blockDim.x + threadIdx.x;
    if (row >= NNc) return;
    int b = row / Lc, i = row % Lc;
    float vi = (chainM[row] * mask[row] + 1e-4f) * fabsf(drand[row]);
    int r = 0;
    for (int j = 0; j < Lc; ++j) {
        int jr = b * Lc + j;
        float vj = (chainM[jr] * mask[jr] + 1e-4f) * fabsf(drand[jr]);
        r += (vj < vi) || (vj == vi && j < i);
    }
    rank[row] = r;
}

__global__ void hes_kernel(const float* __restrict__ hE, const float* __restrict__ hS,
                           const int* __restrict__ Eidx, float* __restrict__ out)
{
    int e = blockIdx.x, h = threadIdx.x;         // 256 threads
    int node = e / Kc, b = node / Lc;
    int jg = b * Lc + Eidx[e];
    out[(size_t)e * 256 + h] =
        (h < 128) ? hE[(size_t)e * Hc + h] : hS[(size_t)jg * Hc + (h - 128)];
}

__global__ void hexv_kernel(const float* __restrict__ hE, const float* __restrict__ hVenc,
                            const int* __restrict__ Eidx, const int* __restrict__ rank,
                            const float* __restrict__ mask, float* __restrict__ out)
{
    int e = blockIdx.x, h = threadIdx.x;         // 384 threads
    int node = e / Kc, b = node / Lc;
    int jg = b * Lc + Eidx[e];
    float fw = mask[node] * ((rank[jg] < rank[node]) ? 0.f : 1.f);
    float v;
    if (h < 128)      v = fw * hE[(size_t)e * Hc + h];
    else if (h < 256) v = 0.f;
    else              v = fw * hVenc[(size_t)jg * Hc + (h - 256)];
    out[(size_t)e * 384 + h] = v;
}

__global__ void dechev_kernel(const float* __restrict__ hV, const float* __restrict__ hES,
                              const float* __restrict__ hEXV, const int* __restrict__ Eidx,
                              const int* __restrict__ rank, const float* __restrict__ mask,
                              float* __restrict__ out)
{
    int e = blockIdx.x, h = threadIdx.x;         // 512 threads
    int node = e / Kc, b = node / Lc;
    int jg = b * Lc + Eidx[e];
    float val;
    if (h < 128) {
        val = hV[(size_t)node * Hc + h];
    } else {
        float bw  = mask[node] * ((rank[jg] < rank[node]) ? 1.f : 0.f);
        float src = (h < 384) ? hES[(size_t)e * 256 + (h - 128)]
                              : hV[(size_t)jg * Hc + (h - 384)];
        val = bw * src + hEXV[(size_t)e * 384 + (h - 128)];
    }
    out[(size_t)e * 512 + h] = val;
}

// ---------------- final attention head ----------------
__global__ void conv_kernel(const float* __restrict__ hV, const float* __restrict__ w,
                            const float* __restrict__ bc, const float* __restrict__ mask,
                            float* __restrict__ out, int domask)
{
    int idx = blockIdx.x * blockDim.x + threadIdx.x;   // (b*H + h)*L + t
    if (idx >= Bc * Hc * Lc) return;
    int t = idx % Lc, h = (idx / Lc) % Hc, b = idx / (Lc * Hc);
    float acc = bc[h];
    for (int u = 0; u < 9; ++u) {
        int tt = t + u - 4;
        if (tt < 0 || tt >= Lc) continue;
        const float* xp = hV + (size_t)(b * Lc + tt) * Hc;
        const float* wp = w + (size_t)h * Hc * 9 + u;
        for (int c = 0; c < Hc; ++c) acc += xp[c] * wp[(size_t)c * 9];
    }
    if (domask && mask[b * Lc + t] == 0.f) acc = -1e9f;
    out[idx] = acc;
}

__global__ void smpool_kernel(const float* __restrict__ o, const float* __restrict__ lg,
                              float* __restrict__ gf)
{
    int wave = threadIdx.x >> 5, lane = threadIdx.x & 31;
    int r = blockIdx.x * 8 + wave;               // b*H + h
    if (r >= Bc * Hc) return;
    const float* lp = lg + (size_t)r * Lc;
    const float* op = o + (size_t)r * Lc;
    float mx = -1e30f;
    for (int t = lane; t < Lc; t += 32) mx = fmaxf(mx, lp[t]);
    for (int off = 16; off >= 1; off >>= 1) mx = fmaxf(mx, __shfl_xor(mx, off, 32));
    float se = 0.f, wsum = 0.f;
    for (int t = lane; t < Lc; t += 32) {
        float e = __expf(lp[t] - mx);
        se += e; wsum += e * op[t];
    }
    for (int off = 16; off >= 1; off >>= 1) {
        se += __shfl_xor(se, off, 32);
        wsum += __shfl_xor(wsum, off, 32);
    }
    if (lane == 0) gf[r] = wsum / se;
}

__global__ void head3_kernel(const float* __restrict__ h2, const float* __restrict__ w,
                             const float* __restrict__ bs, float* __restrict__ out)
{
    int b = threadIdx.x;
    if (b >= Bc) return;
    float acc = bs[0];
    for (int j = 0; j < 64; ++j) acc += h2[b * 64 + j] * w[j];
    out[b] = acc;
}

// =====================================================================
// host orchestration
// =====================================================================
extern "C" void kernel_launch(void* const* d_in, const int* in_sizes, int n_in,
                              void* d_out, int out_size, void* d_ws, size_t ws_size,
                              hipStream_t stream)
{
    (void)in_sizes; (void)n_in; (void)out_size; (void)ws_size;
    auto F  = [&](int i) { return (const float*)d_in[i]; };
    auto Ip = [&](int i) { return (const int*)d_in[i]; };

    const float* X      = F(0);
    const int*   aa     = Ip(1);
    const float* mask   = F(2);
    const float* chainM = F(3);
    const int*   residx = Ip(4);
    const int*   chain  = Ip(5);
    const float* drand  = F(6);
    // params: jax pytree (sorted keys) flattening
    const float* We_b = F(7); const float* We_w = F(8); const float* Ws = F(9);
    auto DEC = [&](int d, int o) { return F(10 + d * 14 + o); };  // W1b,W1w,W2b,W2w,W3b,W3w,dWib,dWiw,dWob,dWow,n1b,n1s,n2b,n2s
    auto ENC = [&](int e, int o) { return F(52 + e * 22 + o); };  // W1b,W1w,W11b,W11w,W12b,W12w,W13b,W13w,W2b,W2w,W3b,W3w,dWib,dWiw,dWob,dWow,n1b,n1s,n2b,n2s,n3b,n3s
    const float* edge_w = F(118);
    const float* en_b = F(119); const float* en_s = F(120);
    const float* pos_b = F(121); const float* pos_w = F(122);
    const float* l1b = F(123); const float* l1w = F(124);
    const float* l2b = F(125); const float* l2w = F(126);
    const float* l3b = F(127); const float* l3w = F(128);
    const float* la_ab = F(129); const float* la_aw = F(130);
    const float* la_fb = F(131); const float* la_fw = F(132);

    float* ws = (float*)d_ws;
    int* Eidx = (int*)(ws + o_Eidx);
    int* rank = (int*)(ws + o_rank);

    auto gemm = [&](const float* Xp, const float* Wp, const float* bp, float* Yp,
                    int M, int N, int Kd, int act) {
        dim3 g((unsigned)((M + 63) / 64), (unsigned)(N / 32));
        gemm_wmma_kernel<<<g, 128, 0, stream>>>(Xp, Wp, bp, Yp, M, N, Kd, act);
    };
    auto ln = [&](const float* Xp, const float* rp, const float* g_, const float* b_,
                  const float* mk, float* Yp, int rows) {
        ln_kernel<<<(rows + 7) / 8, 256, 0, stream>>>(Xp, rp, g_, b_, mk, Yp, rows);
    };

    // ---- features ----
    geom_kernel<<<(NNc + 255) / 256, 256, 0, stream>>>(X, ws + o_atoms);
    topk_kernel<<<NNc, 256, 0, stream>>>(ws + o_atoms, mask, Eidx, ws + o_Dnb);
    matt_kernel<<<(NEc + 255) / 256, 256, 0, stream>>>(mask, Eidx, ws + o_matt);
    edgefeat_kernel<<<(NEc + 127) / 128, 128, 0, stream>>>(
        ws + o_atoms, Eidx, ws + o_Dnb, residx, chain, pos_w, pos_b, ws + o_big);
    gemm(ws + o_big, edge_w, nullptr, ws + o_m2, NEc, 128, EDGE_IN, 0);
    ln(ws + o_m2, nullptr, en_s, en_b, nullptr, ws + o_m1, NEc);
    gemm(ws + o_m1, We_w, We_b, ws + o_hE, NEc, 128, 128, 0);
    zero_kernel<<<(NNc * Hc + 255) / 256, 256, 0, stream>>>(ws + o_hV, NNc * Hc);

    // ---- encoder ----
    for (int l = 0; l < 3; ++l) {
        concat3_kernel<<<NEc, 384, 0, stream>>>(ws + o_hV, ws + o_hE, Eidx, ws + o_big);
        gemm(ws + o_big, ENC(l, 1), ENC(l, 0), ws + o_m1, NEc, 128, 384, 1);   // W1+gelu
        gemm(ws + o_m1, ENC(l, 9), ENC(l, 8), ws + o_m2, NEc, 128, 128, 1);    // W2+gelu
        gemm(ws + o_m2, ENC(l, 11), ENC(l, 10), ws + o_m1, NEc, 128, 128, 0);  // W3
        msgred_kernel<<<NNc, 128, 0, stream>>>(ws + o_m1, ws + o_matt, ws + o_hV, ws + o_nodeT);
        ln(ws + o_nodeT, nullptr, ENC(l, 17), ENC(l, 16), nullptr, ws + o_hV, NNc);      // n1
        gemm(ws + o_hV, ENC(l, 13), ENC(l, 12), ws + o_ffn, NNc, 512, 128, 1); // dWi+gelu
        gemm(ws + o_ffn, ENC(l, 15), ENC(l, 14), ws + o_nodeT, NNc, 128, 512, 0); // dWo
        ln(ws + o_nodeT, ws + o_hV, ENC(l, 19), ENC(l, 18), mask, ws + o_hV, NNc);       // n2 + mask
        concat3_kernel<<<NEc, 384, 0, stream>>>(ws + o_hV, ws + o_hE, Eidx, ws + o_big);
        gemm(ws + o_big, ENC(l, 3), ENC(l, 2), ws + o_m1, NEc, 128, 384, 1);   // W11+gelu
        gemm(ws + o_m1, ENC(l, 5), ENC(l, 4), ws + o_m2, NEc, 128, 128, 1);    // W12+gelu
        gemm(ws + o_m2, ENC(l, 7), ENC(l, 6), ws + o_m1, NEc, 128, 128, 0);    // W13
        ln(ws + o_m1, ws + o_hE, ENC(l, 21), ENC(l, 20), nullptr, ws + o_hE, NEc);       // n3
    }

    // ---- decoder setup ----
    hipMemcpyAsync(ws + o_hVenc, ws + o_hV, sizeof(float) * NNc * Hc,
                   hipMemcpyDeviceToDevice, stream);
    hs_kernel<<<NNc, 128, 0, stream>>>(aa, Ws, ws + o_hS);
    rank_kernel<<<(NNc + 255) / 256, 256, 0, stream>>>(mask, chainM, drand, rank);
    hes_kernel<<<NEc, 256, 0, stream>>>(ws + o_hE, ws + o_hS, Eidx, ws + o_hES);
    hexv_kernel<<<NEc, 384, 0, stream>>>(ws + o_hE, ws + o_hVenc, Eidx, rank, mask, ws + o_hEXV);

    // ---- decoder ----
    for (int l = 0; l < 3; ++l) {
        dechev_kernel<<<NEc, 512, 0, stream>>>(ws + o_hV, ws + o_hES, ws + o_hEXV,
                                               Eidx, rank, mask, ws + o_big);
        gemm(ws + o_big, DEC(l, 1), DEC(l, 0), ws + o_m1, NEc, 128, 512, 1);   // W1+gelu
        gemm(ws + o_m1, DEC(l, 3), DEC(l, 2), ws + o_m2, NEc, 128, 128, 1);    // W2+gelu
        gemm(ws + o_m2, DEC(l, 5), DEC(l, 4), ws + o_m1, NEc, 128, 128, 0);    // W3
        msgred_kernel<<<NNc, 128, 0, stream>>>(ws + o_m1, nullptr, ws + o_hV, ws + o_nodeT);
        ln(ws + o_nodeT, nullptr, DEC(l, 11), DEC(l, 10), nullptr, ws + o_hV, NNc);      // n1
        gemm(ws + o_hV, DEC(l, 7), DEC(l, 6), ws + o_ffn, NNc, 512, 128, 1);   // dWi+gelu
        gemm(ws + o_ffn, DEC(l, 9), DEC(l, 8), ws + o_nodeT, NNc, 128, 512, 0); // dWo
        ln(ws + o_nodeT, ws + o_hV, DEC(l, 13), DEC(l, 12), mask, ws + o_hV, NNc);       // n2 + mask
    }

    // ---- attention-pooled head ----
    conv_kernel<<<(Bc * Hc * Lc + 255) / 256, 256, 0, stream>>>(
        ws + o_hV, la_fw, la_fb, mask, ws + o_o, 0);
    conv_kernel<<<(Bc * Hc * Lc + 255) / 256, 256, 0, stream>>>(
        ws + o_hV, la_aw, la_ab, mask, ws + o_lg, 1);
    smpool_kernel<<<(Bc * Hc + 7) / 8, 256, 0, stream>>>(ws + o_o, ws + o_lg, ws + o_gf);
    gemm(ws + o_gf, l1w, l1b, ws + o_h1, Bc, 128, 128, 1);
    gemm(ws + o_h1, l2w, l2b, ws + o_h2, Bc, 64, 128, 1);
    head3_kernel<<<1, 32, 0, stream>>>(ws + o_h2, l3w, l3b, (float*)d_out);
}
